// ProbabilisticGraphModel_90417651515627
// MI455X (gfx1250) — matrix operand order repaired and checked by reading, not compiled, not run on hardware
//
#include <hip/hip_runtime.h>
#include <hip/hip_bf16.h>

typedef __attribute__((ext_vector_type(16))) _Float16 v16h;
typedef __attribute__((ext_vector_type(8)))  float    v8f;

#define TPB 256

// ---------------- utility kernels ----------------

__global__ void k_zero_f32(float* __restrict__ p, int n) {
    int i = blockIdx.x * blockDim.x + threadIdx.x;
    if (i < n) p[i] = 0.0f;
}

__global__ void k_f32_to_f16(const float* __restrict__ s, _Float16* __restrict__ d, int n) {
    int i = blockIdx.x * blockDim.x + threadIdx.x;
    if (i < n) d[i] = (_Float16)s[i];
}

// deg[dst] += 1 per edge (self-loop added analytically in k_dinv)
__global__ void k_degree(const int* __restrict__ ei, float* __restrict__ deg, int E) {
    int e = blockIdx.x * blockDim.x + threadIdx.x;
    if (e < E) atomicAdd(&deg[ei[E + e]], 1.0f);
}

__global__ void k_dinv(const float* __restrict__ deg, float* __restrict__ dinv, int n) {
    int i = blockIdx.x * blockDim.x + threadIdx.x;
    if (i < n) dinv[i] = rsqrtf(deg[i] + 1.0f);   // +1 = self loop
}

// agg[i][:] = H[i][:] * dinv[i]^2   (self-loop term initializes the accumulator)
__global__ void k_selfloop(const float4* __restrict__ H, const float* __restrict__ dinv,
                           float4* __restrict__ agg, int n, int C4) {
    int t = blockIdx.x * blockDim.x + threadIdx.x;
    if (t >= n * C4) return;
    int i = t / C4, q = t - i * C4;
    float s = dinv[i]; s *= s;
    float4 v = H[(size_t)i * C4 + q];
    v.x *= s; v.y *= s; v.z *= s; v.w *= s;
    agg[(size_t)i * C4 + q] = v;
}

// agg[dst] += dinv[src]*dinv[dst] * H[src]   (one float4 chunk per thread)
__global__ void k_edge_agg(const int* __restrict__ ei, const float* __restrict__ dinv,
                           const float4* __restrict__ H, float* __restrict__ agg,
                           int E, int C4) {
    int t = blockIdx.x * blockDim.x + threadIdx.x;
    if (t >= E * C4) return;
    int e = t / C4, q = t - e * C4;
    int s = ei[e], d = ei[E + e];
    float nrm = dinv[s] * dinv[d];
    float4 v = H[(size_t)s * C4 + q];
    float* o = agg + ((size_t)d * C4 + q) * 4;
    atomicAdd(o + 0, v.x * nrm);
    atomicAdd(o + 1, v.y * nrm);
    atomicAdd(o + 2, v.z * nrm);
    atomicAdd(o + 3, v.w * nrm);
}

// ---------------- WMMA GEMM: D[M,N] = act(A[M,K] @ B[K,N] + bias) ----------------
// One wave computes one 16x16 output tile; K/N compile-time so the k-loop
// fully unrolls into chained v_wmma_f32_16x16x32_f16 ops.
template <int K, int N>
__global__ void k_gemm_wmma(const float* __restrict__ A, const _Float16* __restrict__ B,
                            const float* __restrict__ bias, float* __restrict__ D,
                            int M, int relu) {
    const int lane = threadIdx.x & 31;
    const int wave = threadIdx.x >> 5;
    const int tilesN = N >> 4;
    const int totalTiles = (M >> 4) * tilesN;
    int tile = blockIdx.x * 8 + wave;
    if (tile >= totalTiles) return;           // wave-uniform: EXEC stays all-1s for WMMA
    const int tm = tile / tilesN, tn = tile - tm * tilesN;
    const int r    = lane & 15;               // A-row / B-col / D-col within tile
    const int half = lane >> 4;

    v8f c = {};
    const float* Arow = A + (size_t)(tm * 16 + r) * K;
#pragma unroll
    for (int k0 = 0; k0 < K; k0 += 32) {
        v16h a, b;
        // A fragment: 16-bit A-matrix 16x32 layout (regs 0-3: K=half*8+0..7, regs 4-7: +16)
#pragma unroll
        for (int j = 0; j < 8; ++j) {
            int kb = (j < 4) ? (k0 + half * 8 + 2 * j)
                             : (k0 + 16 + half * 8 + 2 * (j - 4));
            a[2 * j]     = (_Float16)Arow[kb];
            a[2 * j + 1] = (_Float16)Arow[kb + 1];
        }
        // B fragment: lane = column, regs 0-3: K=half*8+0..7, regs 4-7: +16 (matches A)
        const _Float16* Bp = B + (size_t)(k0 + half * 8) * N + (tn * 16 + r);
#pragma unroll
        for (int j = 0; j < 8; ++j) {
            int kk = (j < 4) ? (2 * j) : (16 + 2 * (j - 4));
            b[2 * j]     = Bp[kk * N];
            b[2 * j + 1] = Bp[(kk + 1) * N];
        }
        c = __builtin_amdgcn_wmma_f32_16x16x32_f16(
                /*neg_a=*/false, a, /*neg_b=*/false, b,
                /*c_mod=*/(short)0, c, /*reuse_a=*/false, /*reuse_b=*/false);
    }
    const int col = tn * 16 + r;
    const float bv = bias ? bias[col] : 0.0f;
#pragma unroll
    for (int m = 0; m < 8; ++m) {            // D: reg m -> row m + half*8, col = lane%16
        int row = tm * 16 + m + half * 8;
        float v = c[m] + bv;
        if (relu) v = fmaxf(v, 0.0f);
        D[(size_t)row * N + col] = v;
    }
}

// ---------------- pooling / head ----------------

__global__ void k_count(const int* __restrict__ batch, float* __restrict__ cnt, int n) {
    int i = blockIdx.x * blockDim.x + threadIdx.x;
    if (i < n) atomicAdd(&cnt[batch[i]], 1.0f);
}

// pooled[g] += relu(agg2[i] + b2)   (bias+relu fused into the pooling scatter)
__global__ void k_pool(const float4* __restrict__ agg, const float* __restrict__ b2,
                       const int* __restrict__ batch, float* __restrict__ pooled,
                       int n, int C4) {
    int t = blockIdx.x * blockDim.x + threadIdx.x;
    if (t >= n * C4) return;
    int i = t / C4, q = t - i * C4;
    int g = batch[i];
    float4 v  = agg[(size_t)i * C4 + q];
    float4 bb = ((const float4*)b2)[q];
    float* o = pooled + ((size_t)g * C4 + q) * 4;
    atomicAdd(o + 0, fmaxf(v.x + bb.x, 0.0f));
    atomicAdd(o + 1, fmaxf(v.y + bb.y, 0.0f));
    atomicAdd(o + 2, fmaxf(v.z + bb.z, 0.0f));
    atomicAdd(o + 3, fmaxf(v.w + bb.w, 0.0f));
}

// out[g] = (pooled[g]/max(cnt,1)) . fcW + fcb    (one block of 64 threads per graph)
__global__ void k_final(const float* __restrict__ pooled, const float* __restrict__ cnt,
                        const float* __restrict__ fcW, const float* __restrict__ fcb,
                        float* __restrict__ out) {
    __shared__ float red[64];
    int g = blockIdx.x, t = threadIdx.x;
    float c = fmaxf(cnt[g], 1.0f);
    red[t] = (pooled[(size_t)g * 64 + t] / c) * fcW[t];
    __syncthreads();
    for (int s = 32; s > 0; s >>= 1) {
        if (t < s) red[t] += red[t + s];
        __syncthreads();
    }
    if (t == 0) out[g] = red[0] + fcb[0];
}

// ---------------- launch ----------------

static inline int cdiv(int a, int b) { return (a + b - 1) / b; }

extern "C" void kernel_launch(void* const* d_in, const int* in_sizes, int n_in,
                              void* d_out, int out_size, void* d_ws, size_t ws_size,
                              hipStream_t stream) {
    const float* x    = (const float*)d_in[0];   // [N,64]
    const int*   ei   = (const int*)  d_in[1];   // [2,E]
    const int*   batch= (const int*)  d_in[2];   // [N]
    const float* W1   = (const float*)d_in[3];   // [64,128]
    const float* b1   = (const float*)d_in[4];   // [128]
    const float* W2   = (const float*)d_in[5];   // [128,64]
    const float* b2   = (const float*)d_in[6];   // [64]
    const float* fcW  = (const float*)d_in[7];   // [64,1]
    const float* fcb  = (const float*)d_in[8];   // [1]
    float* out = (float*)d_out;

    const int N = in_sizes[0] / 64;
    const int E = in_sizes[1] / 2;
    const int G = out_size;

    // workspace carve-out (256B aligned)
    char* ws = (char*)d_ws;
    size_t off = 0;
    auto carve = [&](size_t bytes) -> char* {
        char* p = ws + off;
        off = (off + bytes + 255) & ~(size_t)255;
        return p;
    };
    float*    deg   = (float*)   carve((size_t)N * 4);
    float*    dinv  = (float*)   carve((size_t)N * 4);
    float*    aggA  = (float*)   carve((size_t)N * 64 * 4);   // agg1, reused as agg2
    float*    h1    = (float*)   carve((size_t)N * 128 * 4);
    float*    t2    = (float*)   carve((size_t)N * 64 * 4);
    _Float16* W1h   = (_Float16*)carve((size_t)64 * 128 * 2);
    _Float16* W2h   = (_Float16*)carve((size_t)128 * 64 * 2);
    float*    pooled= (float*)   carve((size_t)G * 64 * 4);
    float*    cnt   = (float*)   carve((size_t)G * 4);
    (void)ws_size; (void)n_in;

    // 0) zero accumulators (every call: graph-replay safe)
    k_zero_f32<<<cdiv(N, TPB),       TPB, 0, stream>>>(deg, N);
    k_zero_f32<<<cdiv(G * 64, TPB),  TPB, 0, stream>>>(pooled, G * 64);
    k_zero_f32<<<cdiv(G, TPB),       TPB, 0, stream>>>(cnt, G);

    // 1) degrees + symmetric norm
    k_degree<<<cdiv(E, TPB), TPB, 0, stream>>>(ei, deg, E);
    k_dinv  <<<cdiv(N, TPB), TPB, 0, stream>>>(deg, dinv, N);

    // 2) weights -> f16 once per call
    k_f32_to_f16<<<cdiv(64 * 128, TPB), TPB, 0, stream>>>(W1, W1h, 64 * 128);
    k_f32_to_f16<<<cdiv(128 * 64, TPB), TPB, 0, stream>>>(W2, W2h, 128 * 64);

    // 3) layer 1: aggregate-then-transform (edge traffic = 64 floats, not 128)
    k_selfloop<<<cdiv(N * 16, TPB), TPB, 0, stream>>>((const float4*)x, dinv,
                                                      (float4*)aggA, N, 16);
    k_edge_agg<<<cdiv(E * 16, TPB), TPB, 0, stream>>>(ei, dinv, (const float4*)x,
                                                      aggA, E, 16);
    {   // h1 = relu(aggA @ W1 + b1)   [N,64]@[64,128]
        int tiles = (N >> 4) * (128 >> 4);
        k_gemm_wmma<64, 128><<<cdiv(tiles, 8), TPB, 0, stream>>>(aggA, W1h, b1, h1, N, 1);
    }

    // 4) layer 2: transform-then-aggregate (64 output cols < 128 input cols)
    {   // t2 = h1 @ W2   [N,128]@[128,64]
        int tiles = (N >> 4) * (64 >> 4);
        k_gemm_wmma<128, 64><<<cdiv(tiles, 8), TPB, 0, stream>>>(h1, W2h, nullptr, t2, N, 0);
    }
    k_selfloop<<<cdiv(N * 16, TPB), TPB, 0, stream>>>((const float4*)t2, dinv,
                                                      (float4*)aggA, N, 16);
    k_edge_agg<<<cdiv(E * 16, TPB), TPB, 0, stream>>>(ei, dinv, (const float4*)t2,
                                                      aggA, E, 16);

    // 5) mean-pool (bias+relu fused) + linear head
    k_count<<<cdiv(N, TPB), TPB, 0, stream>>>(batch, cnt, N);
    k_pool <<<cdiv(N * 16, TPB), TPB, 0, stream>>>((const float4*)aggA, b2, batch,
                                                   pooled, N, 16);
    k_final<<<G, 64, 0, stream>>>(pooled, cnt, fcW, fcb, out);
}
